// NonLocalDenoise_65481071395174
// MI455X (gfx1250) — matrix operand order repaired and checked by reading, not compiled, not run on hardware
//
#include <hip/hip_runtime.h>
#include <hip/hip_fp16.h>
#include <stdint.h>

typedef __attribute__((ext_vector_type(16))) _Float16 v16h;
typedef __attribute__((ext_vector_type(8)))  _Float16 v8h;
typedef __attribute__((ext_vector_type(8)))  float    v8f;
typedef __attribute__((ext_vector_type(4)))  float    v4f;

#define N_  4
#define C_  128
#define HW_ 9216
#define CE_ 64
#define LOG2E 1.4426950408889634f

union V16U { v16h v; v8h h[2]; };
union V8FU { v8f v; v4f q[2]; };

__device__ __forceinline__ v16h combine16(v8h lo, v8h hi) {
    V16U u; u.h[0] = lo; u.h[1] = hi; return u.v;
}

// Fused single-instruction DPP16 butterfly steps: v_max_num_f32_dpp /
// v_add_f32_dpp with row_ror (reduces within each 16-lane row). Avoids the
// mov_dpp + canonicalize pair the intrinsic path produces.
#define RORMAX(x, AMT) ({ float _d;                                            \
    asm("v_max_num_f32_dpp %0, %1, %1 row_ror:" AMT                            \
        " row_mask:0xf bank_mask:0xf bound_ctrl:1"                             \
        : "=v"(_d) : "v"(x)); _d; })
#define RORADD(x, AMT) ({ float _d;                                            \
    asm("v_add_f32_dpp %0, %1, %1 row_ror:" AMT                                \
        " row_mask:0xf bank_mask:0xf bound_ctrl:1"                             \
        : "=v"(_d) : "v"(x)); _d; })

__device__ __forceinline__ float rmax16(float x) {
    x = RORMAX(x, "8");
    x = RORMAX(x, "4");
    x = RORMAX(x, "2");
    x = RORMAX(x, "1");
    return x;
}
__device__ __forceinline__ float rsum16(float x) {
    x = RORADD(x, "8");
    x = RORADD(x, "4");
    x = RORADD(x, "2");
    x = RORADD(x, "1");
    return x;
}

// ---------------------------------------------------------------------------
// Embed: e = PReLU(W x + b); q,k as (N*HW, 64) fp16, v as (N, C, HW) fp16.
// ---------------------------------------------------------------------------
__global__ __launch_bounds__(256) void embed_kernel(
    const float* __restrict__ x,  const float* __restrict__ w1,
    const float* __restrict__ b1, const float* __restrict__ a1,
    const float* __restrict__ w2, const float* __restrict__ b2,
    const float* __restrict__ a2,
    _Float16* __restrict__ qh, _Float16* __restrict__ kh, _Float16* __restrict__ vh)
{
    __shared__ float w1s[C_ * CE_];   // transposed [c][o]
    __shared__ float w2s[C_ * CE_];
    const int t = threadIdx.x;
    for (int idx = t; idx < CE_ * C_; idx += 256) {
        const int o = idx >> 7, c = idx & 127;
        w1s[c * CE_ + o] = w1[idx];
        w2s[c * CE_ + o] = w2[idx];
    }
    __syncthreads();

    const int p = blockIdx.x * 256 + t;
    const int n = p / HW_, i = p % HW_;
    const float* xp = x  + (size_t)n * C_ * HW_ + i;
    _Float16*    vp = vh + (size_t)n * C_ * HW_ + i;
    const float s1 = a1[0], s2 = a2[0];

    float acc[CE_];
    #pragma unroll
    for (int o = 0; o < CE_; ++o) acc[o] = 0.f;
    for (int c = 0; c < C_; ++c) {
        const float xc = xp[(size_t)c * HW_];
        vp[(size_t)c * HW_] = (_Float16)xc;
        #pragma unroll
        for (int o = 0; o < CE_; ++o) acc[o] = fmaf(w1s[c * CE_ + o], xc, acc[o]);
    }
    _Float16* qp = qh + (size_t)p * CE_;
    #pragma unroll
    for (int o = 0; o < CE_; ++o) {
        const float e = acc[o] + b1[o];
        qp[o] = (_Float16)(e > 0.f ? e : s1 * e);
    }

    #pragma unroll
    for (int o = 0; o < CE_; ++o) acc[o] = 0.f;
    for (int c = 0; c < C_; ++c) {
        const float xc = xp[(size_t)c * HW_];
        #pragma unroll
        for (int o = 0; o < CE_; ++o) acc[o] = fmaf(w2s[c * CE_ + o], xc, acc[o]);
    }
    _Float16* kp = kh + (size_t)p * CE_;
    #pragma unroll
    for (int o = 0; o < CE_; ++o) {
        const float e = acc[o] + b2[o];
        kp[o] = (_Float16)(e > 0.f ? e : s2 * e);
    }
}

// ---------------------------------------------------------------------------
// Fused flash attention. 8 waves/block share K/V tiles staged in LDS via
// double-buffered global_load_async_to_lds_b128 (ASYNCcnt); each wave owns a
// 16-row strip. Softmax in exp2 domain (raw v_exp_f32, fused DPP reductions).
// ---------------------------------------------------------------------------
__global__ __launch_bounds__(256) void attn_kernel(
    const _Float16* __restrict__ q, const _Float16* __restrict__ k,
    const _Float16* __restrict__ v, float* __restrict__ out)
{
    __shared__ __align__(16) _Float16 ktile[2][32 * CE_];   // 2 x 4 KB
    __shared__ __align__(16) _Float16 vtile[2][C_ * 32];    // 2 x 8 KB
    __shared__ __align__(16) _Float16 plds_all[8][16 * 32]; // 8 x 1 KB

    const int t    = threadIdx.x;
    const int lane = t & 31;
    const int wave = t >> 5;
    _Float16* plds = plds_all[wave];

    const int strip = blockIdx.x * 8 + wave;   // 8 strips/block, same batch n
    const int n  = strip / (HW_ / 16);
    const int i0 = (strip % (HW_ / 16)) * 16;

    const _Float16* qb = q + (size_t)n * HW_ * CE_;
    const uint64_t  kb = (uint64_t)(k + (size_t)n * HW_ * CE_);
    const uint64_t  vb = (uint64_t)(v + (size_t)n * C_ * HW_);
    float*          ob = out + (size_t)n * C_ * HW_;

    const int mrow   = lane & 15;
    const int khalf  = (lane >> 4) << 3;   // A-frag K offset (0/8)
    const int kb16   = (lane >> 4) << 4;   // K offset 0/16, unused (B-frags use khalf)
    const int rowoff = (lane >> 4) << 3;   // C/D-frag row offset (0/8)

    // Cooperative async tile fetch: K 32x64 (256 x 16B), V 128x32 (512 x 16B).
    const unsigned k_jrow = t >> 3, k_c8 = (t & 7) * 8;
    const unsigned k_lds0 = (unsigned)(uintptr_t)&ktile[0][k_jrow * CE_ + k_c8];
    const unsigned k_lds1 = (unsigned)(uintptr_t)&ktile[1][k_jrow * CE_ + k_c8];
    unsigned v_lds0[2], v_lds1[2], v_c[2], v_j8[2];
    #pragma unroll
    for (int r = 0; r < 2; ++r) {
        const unsigned idx = r * 256 + t;
        v_c[r] = idx >> 2; v_j8[r] = (idx & 3) * 8;
        v_lds0[r] = (unsigned)(uintptr_t)&vtile[0][v_c[r] * 32 + v_j8[r]];
        v_lds1[r] = (unsigned)(uintptr_t)&vtile[1][v_c[r] * 32 + v_j8[r]];
    }
    auto issue_tile = [&](int buf, int j0) {
        const unsigned kgo = (unsigned)(((unsigned)(j0 + k_jrow) * CE_ + k_c8) * 2u);
        const unsigned kl  = buf ? k_lds1 : k_lds0;
        asm volatile("global_load_async_to_lds_b128 %0, %1, %2"
                     :: "v"(kl), "v"(kgo), "s"(kb) : "memory");
        #pragma unroll
        for (int r = 0; r < 2; ++r) {
            const unsigned vgo = (unsigned)((v_c[r] * HW_ + (unsigned)j0 + v_j8[r]) * 2u);
            const unsigned vl  = buf ? v_lds1[r] : v_lds0[r];
            asm volatile("global_load_async_to_lds_b128 %0, %1, %2"
                         :: "v"(vl), "v"(vgo), "s"(vb) : "memory");
        }
    };

    // Q fragments (16x64), resident for the whole strip
    v16h qf[2];
    #pragma unroll
    for (int ks = 0; ks < 2; ++ks) {
        const _Float16* p = qb + (size_t)(i0 + mrow) * CE_ + ks * 32;
        qf[ks] = combine16(*(const v8h*)(p + khalf), *(const v8h*)(p + 16 + khalf));
    }

    v8f oacc[8];
    #pragma unroll
    for (int cf = 0; cf < 8; ++cf) oacc[cf] = (v8f){};
    float mrun[8], lrun[8];   // mrun kept in log2 domain
    #pragma unroll
    for (int r = 0; r < 8; ++r) { mrun[r] = -__builtin_inff(); lrun[r] = 0.f; }

    issue_tile(0, 0);
    asm volatile("s_wait_asynccnt 0x0" ::: "memory");
    __syncthreads();

    const int NT = HW_ / 32;
    for (int jt = 0; jt < NT; ++jt) {
        const int cur = jt & 1;
        if (jt + 1 < NT) issue_tile(cur ^ 1, (jt + 1) * 32);  // overlap DMA
        const _Float16* kt = ktile[cur];
        const _Float16* vt = vtile[cur];

        // ---- S(16x32) = Q(16x64) * K^T: preload all B-frags, then 4 WMMAs ----
        v16h bf[2][2];
        #pragma unroll
        for (int jt2 = 0; jt2 < 2; ++jt2) {
            const _Float16* kp = kt + (jt2 * 16 + mrow) * CE_ + khalf;
            #pragma unroll
            for (int ks = 0; ks < 2; ++ks)
                bf[jt2][ks] = combine16(*(const v8h*)(kp + ks * 32),
                                        *(const v8h*)(kp + ks * 32 + 16));
        }
        v8f sf[2];
        #pragma unroll
        for (int jt2 = 0; jt2 < 2; ++jt2) {
            sf[jt2] = (v8f){};
            #pragma unroll
            for (int ks = 0; ks < 2; ++ks)
                sf[jt2] = __builtin_amdgcn_wmma_f32_16x16x32_f16(
                    false, qf[ks], false, bf[jt2][ks], (short)0, sf[jt2], false, false);
        }
        // scale to log2 domain once; all exps below are raw v_exp_f32
        sf[0] *= LOG2E; sf[1] *= LOG2E;

        // ---- online softmax: fused DPP row reductions ----
        float fac[8];
        #pragma unroll
        for (int r = 0; r < 8; ++r) {
            const float mx   = rmax16(fmaxf(sf[0][r], sf[1][r]));
            const float mnew = fmaxf(mrun[r], mx);
            fac[r]  = __builtin_amdgcn_exp2f(mrun[r] - mnew);
            mrun[r] = mnew;
            const float p0 = __builtin_amdgcn_exp2f(sf[0][r] - mnew);
            const float p1 = __builtin_amdgcn_exp2f(sf[1][r] - mnew);
            lrun[r] = lrun[r] * fac[r] + rsum16(p0 + p1);
            plds[(r + rowoff) * 32 + mrow]      = (_Float16)p0;
            plds[(r + rowoff) * 32 + 16 + mrow] = (_Float16)p1;
        }
        // rescale O accumulators (vector: v_pk_mul_f32)
        v8f facv;
        #pragma unroll
        for (int r = 0; r < 8; ++r) facv[r] = fac[r];
        #pragma unroll
        for (int cf = 0; cf < 8; ++cf) oacc[cf] *= facv;

        asm volatile("s_wait_dscnt 0x0" ::: "memory");

        // ---- O += P(16x32) * V(32x128): V frags in groups of 4 ----
        const _Float16* pl = plds + mrow * 32;
        v16h pf = combine16(*(const v8h*)(pl + khalf), *(const v8h*)(pl + 16 + khalf));
        #pragma unroll
        for (int g = 0; g < 2; ++g) {
            v16h vf[4];
            #pragma unroll
            for (int u = 0; u < 4; ++u) {
                const _Float16* vp = vt + ((g * 4 + u) * 16 + mrow) * 32 + khalf;
                vf[u] = combine16(*(const v8h*)vp, *(const v8h*)(vp + 16));
            }
            #pragma unroll
            for (int u = 0; u < 4; ++u)
                oacc[g * 4 + u] = __builtin_amdgcn_wmma_f32_16x16x32_f16(
                    false, pf, false, vf[u], (short)0, oacc[g * 4 + u], false, false);
        }

        asm volatile("s_wait_asynccnt 0x0" ::: "memory");  // next tile landed
        __syncthreads();                                   // all waves done w/ cur
    }

    // ---- normalize and store: 8 consecutive floats/lane -> 2x b128 per tile ----
    v8f lv;
    #pragma unroll
    for (int r = 0; r < 8; ++r) lv[r] = 1.0f / lrun[r];
    #pragma unroll
    for (int cf = 0; cf < 8; ++cf) {
        V8FU o; o.v = oacc[cf] * lv;
        float* dst = ob + (size_t)(cf * 16 + mrow) * HW_ + i0 + rowoff;
        *(v4f*)(dst)     = o.q[0];
        *(v4f*)(dst + 4) = o.q[1];
    }
}

extern "C" void kernel_launch(void* const* d_in, const int* in_sizes, int n_in,
                              void* d_out, int out_size, void* d_ws, size_t ws_size,
                              hipStream_t stream)
{
    (void)in_sizes; (void)n_in; (void)out_size; (void)ws_size;
    const float* x  = (const float*)d_in[0];
    const float* w1 = (const float*)d_in[1];
    const float* b1 = (const float*)d_in[2];
    const float* a1 = (const float*)d_in[3];
    const float* w2 = (const float*)d_in[4];
    const float* b2 = (const float*)d_in[5];
    const float* a2 = (const float*)d_in[6];
    float* out = (float*)d_out;

    char* ws = (char*)d_ws;
    const size_t qk_bytes = (size_t)N_ * HW_ * CE_ * sizeof(_Float16);
    _Float16* qh = (_Float16*)(ws);
    _Float16* kh = (_Float16*)(ws + qk_bytes);
    _Float16* vh = (_Float16*)(ws + 2 * qk_bytes);

    embed_kernel<<<(N_ * HW_) / 256, 256, 0, stream>>>(x, w1, b1, a1, w2, b2, a2,
                                                       qh, kh, vh);
    attn_kernel<<<(N_ * HW_ / 16) / 8, 256, 0, stream>>>(qh, kh, vh, out);
}
